// SubBlock_3255585210533
// MI455X (gfx1250) — matrix operand order repaired and checked by reading, not compiled, not run on hardware
//
#include <hip/hip_runtime.h>
#include <hip/hip_bf16.h>

// ============================================================================
// Equivariant block (a/v/d streams) for MI455X (gfx1250, wave32, WMMA).
// Every heavy op (linear_mix, rank-64 bilinear projections, MLPs, 7-tap chain
// convs) is a GEMM over M = 16384 positions run on v_wmma_f32_16x16x32_bf16.
// This revision:
//   * pre-splits conv kernels (O,I,7) -> 7 contiguous (O,I) matrices so the
//     B-fragment load is always the vectorized b128-clause path (the strided
//     path previously produced per-element scalar loads),
//   * 2x2 tiles per wave (32Mx32N): 4 independent WMMAs per K-step -> fewer
//     hazard NOPs, half the global bytes per WMMA,
//   * K as a template parameter (64/128/256) -> fully unrolled K loops.
//
// Input flattening assumption (depth-first insertion order of setup_inputs):
//  0:x_a 1:x_v 2:x_d, 3..10: mix(Waa,Wad,Wvv,Wvd,Wdd,Wdt,Wda,Wdv),
//  11..37: tp {aa,vv,dd,av,dv,vxv,ad,vov,dmd} x {l,r,o},
//  38..41: gn(a_g,a_b,v_g,d_g), 42..44 conv_a, 45..47 conv_v, 48..50 conv_d,
//  51..53 mlp_a, 54..56 mlp_v, 57..59 mlp_d.
// ============================================================================

typedef __bf16 bf16_t;
typedef __attribute__((ext_vector_type(16))) bf16_t v16bf;
typedef __attribute__((ext_vector_type(8)))  float  v8f;

#define P_TOT  16384      // B*N positions
#define NCHAIN 2048       // chain length (power of two)
#define DA_    256
#define DV_    128
#define DD_    64
#define PA_    (P_TOT*DA_)
#define PV_    (P_TOT*DV_)
#define PD_    (P_TOT*DD_)

// ---------------------------------------------------------------------------
// Fragment loaders per the CDNA5 bf16 WMMA VGPR layouts.
// A (16x32): lane m = l16; elements 0..7  = K[half*8 .. +8),
//                          elements 8..15 = K[16+half*8 .. +8).
// B (32x16): lane n = l16; same K mapping as A (elements 0..7 = K[half*8 .. +8),
//                          elements 8..15 = K[16+half*8 .. +8)).
// ---------------------------------------------------------------------------
__device__ inline v16bf load_a_frag(const float* __restrict__ rowA, int k0,
                                    int half, bool valid) {
  v16bf f;
  if (valid) {
    const float4* q0 = (const float4*)(rowA + k0 + half * 8);
    const float4* q1 = (const float4*)(rowA + k0 + 16 + half * 8);
    float fa[16];
    ((float4*)fa)[0] = q0[0]; ((float4*)fa)[1] = q0[1];
    ((float4*)fa)[2] = q1[0]; ((float4*)fa)[3] = q1[1];
#pragma unroll
    for (int i = 0; i < 16; ++i) f[i] = (bf16_t)fa[i];
  } else {
#pragma unroll
    for (int i = 0; i < 16; ++i) f[i] = (bf16_t)0.0f;
  }
  return f;
}
__device__ inline v16bf load_b_frag(const float* __restrict__ rowW, int k0, int half) {
  const float4* qb = (const float4*)(rowW + k0 + half * 8);
  const float4* qc = (const float4*)(rowW + k0 + 16 + half * 8);
  float fb[16];
  ((float4*)fb)[0] = qb[0]; ((float4*)fb)[1] = qb[1];
  ((float4*)fb)[2] = qc[0]; ((float4*)fb)[3] = qc[1];
  v16bf f;
#pragma unroll
  for (int i = 0; i < 16; ++i) f[i] = (bf16_t)fb[i];
  return f;
}
// C/D layout: lane -> N = l16; VGPR g -> M = half*8 + g
__device__ inline void store_tile(float* __restrict__ C, int ldc, int mbase,
                                  int nbase, int half, int l16, v8f acc,
                                  float alpha, int accum, int act) {
  const int nc = nbase + l16;
#pragma unroll
  for (int g = 0; g < 8; ++g) {
    const int mr = mbase + half * 8 + g;
    const size_t idx = (size_t)mr * ldc + nc;
    float v = alpha * acc[g];
    if (accum) v += C[idx];
    if (act)   v = v > 0.0f ? v : 0.1f * v;
    C[idx] = v;
  }
}

// ---------------------------------------------------------------------------
// WMMA GEMM, 2x2 tiles per wave:  C[m,n] (+)= alpha * sum_k A[m,k]*W[n,k]
//   K is compile-time (64/128/256) -> fully unrolled, 4 independent WMMAs per
//   K-step. Optional row shift (conv tap) zero-masks rows whose shifted index
//   leaves [0, NCHAIN) within a chain. act==1: fused leaky-relu(0.1).
// ---------------------------------------------------------------------------
template <int K>
__global__ __launch_bounds__(256) void k_wmma_gemm(
    const float* __restrict__ A, int lda,
    const float* __restrict__ W, int ldw,
    float* __restrict__ C, int ldc,
    int Nout, float alpha, int accum, int shift, int act)
{
  const int wave = threadIdx.x >> 5;
  const int lane = threadIdx.x & 31;
  const int Mt2 = P_TOT / 32;
  const int Nt2 = Nout >> 5;
  const int tile = blockIdx.x * 8 + wave;
  if (tile >= Mt2 * Nt2) return;
  const int nt2 = tile / Mt2;          // consecutive waves share nt2 -> W reuse
  const int mt2 = tile - nt2 * Mt2;
  const int half = lane >> 4;
  const int l16  = lane & 15;

  const int mb = mt2 * 32;             // M base of the 32x32 macro-tile
  const int nb = nt2 * 32;
  const int r0 = mb + l16;             // A-frag rows (this lane)
  const int r1 = r0 + 16;
  bool valid0 = true, valid1 = true;
  if (shift != 0) {
    const int s0 = (r0 & (NCHAIN - 1)) + shift;
    const int s1 = (r1 & (NCHAIN - 1)) + shift;
    valid0 = (unsigned)s0 < (unsigned)NCHAIN;
    valid1 = (unsigned)s1 < (unsigned)NCHAIN;
  }
  const float* rowA0 = A + (size_t)(r0 + shift) * lda;
  const float* rowA1 = A + (size_t)(r1 + shift) * lda;
  const float* rowW0 = W + (size_t)(nb + l16) * ldw;
  const float* rowW1 = rowW0 + (size_t)16 * ldw;

  if (valid0) __builtin_prefetch(rowA0, 0, 0);   // global_prefetch_b8
  __builtin_prefetch(rowW0, 0, 1);

  v8f acc00, acc01, acc10, acc11;
#pragma unroll
  for (int i = 0; i < 8; ++i) { acc00[i] = 0.f; acc01[i] = 0.f; acc10[i] = 0.f; acc11[i] = 0.f; }

#pragma unroll
  for (int k0 = 0; k0 < K; k0 += 32) {
    const v16bf a0 = load_a_frag(rowA0, k0, half, valid0);
    const v16bf a1 = load_a_frag(rowA1, k0, half, valid1);
    const v16bf b0 = load_b_frag(rowW0, k0, half);
    const v16bf b1 = load_b_frag(rowW1, k0, half);
    acc00 = __builtin_amdgcn_wmma_f32_16x16x32_bf16(false, a0, false, b0, (short)0, acc00, false, false);
    acc01 = __builtin_amdgcn_wmma_f32_16x16x32_bf16(false, a0, false, b1, (short)0, acc01, false, false);
    acc10 = __builtin_amdgcn_wmma_f32_16x16x32_bf16(false, a1, false, b0, (short)0, acc10, false, false);
    acc11 = __builtin_amdgcn_wmma_f32_16x16x32_bf16(false, a1, false, b1, (short)0, acc11, false, false);
  }

  store_tile(C, ldc, mb,      nb,      half, l16, acc00, alpha, accum, act);
  store_tile(C, ldc, mb,      nb + 16, half, l16, acc01, alpha, accum, act);
  store_tile(C, ldc, mb + 16, nb,      half, l16, acc10, alpha, accum, act);
  store_tile(C, ldc, mb + 16, nb + 16, half, l16, acc11, alpha, accum, act);
}

// ------------------------- elementwise / reduction -------------------------
__global__ void k_split_v(const float* __restrict__ xv, float* __restrict__ vp) {
  int i = blockIdx.x * 256 + threadIdx.x;
  if (i >= PV_) return;
  vp[0 * (size_t)PV_ + i] = xv[3 * (size_t)i + 0];
  vp[1 * (size_t)PV_ + i] = xv[3 * (size_t)i + 1];
  vp[2 * (size_t)PV_ + i] = xv[3 * (size_t)i + 2];
}
__global__ void k_split_d(const float* __restrict__ xd, float* __restrict__ dp,
                          float* __restrict__ tr, float* __restrict__ vd) {
  int i = blockIdx.x * 256 + threadIdx.x;
  if (i >= PD_) return;
  float e[9];
#pragma unroll
  for (int q = 0; q < 9; ++q) { e[q] = xd[9 * (size_t)i + q]; dp[q * (size_t)PD_ + i] = e[q]; }
  tr[i] = e[0] + e[4] + e[8];
  vd[0 * (size_t)PD_ + i] = e[5] - e[7];   // eps_{0jk} d_jk
  vd[1 * (size_t)PD_ + i] = e[6] - e[2];
  vd[2 * (size_t)PD_ + i] = e[1] - e[3];
}
// split (O,I,7) conv kernel into 7 contiguous (O,I) matrices
__global__ void k_split_conv(const float* __restrict__ w, float* __restrict__ o, int n /*C*C*/) {
  int i = blockIdx.x * 256 + threadIdx.x;
  if (i >= n * 7) return;
  int t = i % 7, oi = i / 7;
  o[(size_t)t * n + oi] = w[i];
}
// d2[i][j] += eps_ijk t_k   (t_k = Wdv . v_k)
__global__ void k_eps_add(float* __restrict__ d2, const float* __restrict__ t) {
  int i = blockIdx.x * 256 + threadIdx.x;
  if (i >= PD_) return;
  float t0 = t[i], t1 = t[(size_t)PD_ + i], t2 = t[2 * (size_t)PD_ + i];
  d2[1 * (size_t)PD_ + i] += t2;  d2[2 * (size_t)PD_ + i] -= t1;
  d2[3 * (size_t)PD_ + i] -= t2;  d2[5 * (size_t)PD_ + i] += t0;
  d2[6 * (size_t)PD_ + i] += t1;  d2[7 * (size_t)PD_ + i] -= t0;
}
__global__ void k_had(float* __restrict__ c, const float* __restrict__ a,
                      const float* __restrict__ b, int n) {
  int i = blockIdx.x * 256 + threadIdx.x;
  if (i < n) c[i] = a[i] * b[i];
}
__global__ void k_dotn(float* __restrict__ c, const float* __restrict__ a,
                       const float* __restrict__ b, int n, int comps) {
  int i = blockIdx.x * 256 + threadIdx.x;
  if (i >= n) return;
  float s = 0.f;
  for (int t = 0; t < comps; ++t) s += a[(size_t)t * n + i] * b[(size_t)t * n + i];
  c[i] = s;
}
// X_i = sum_j L_{ij} * R_j (planes stride n)
__global__ void k_matvec3(float* __restrict__ x, const float* __restrict__ L,
                          const float* __restrict__ R, int n) {
  int i = blockIdx.x * 256 + threadIdx.x;
  if (i >= n) return;
  float r0 = R[i], r1 = R[(size_t)n + i], r2 = R[2 * (size_t)n + i];
#pragma unroll
  for (int r = 0; r < 3; ++r)
    x[(size_t)r * n + i] = L[(size_t)(r * 3 + 0) * n + i] * r0 +
                           L[(size_t)(r * 3 + 1) * n + i] * r1 +
                           L[(size_t)(r * 3 + 2) * n + i] * r2;
}
__global__ void k_cross3(float* __restrict__ x, const float* __restrict__ a,
                         const float* __restrict__ b, int n) {
  int i = blockIdx.x * 256 + threadIdx.x;
  if (i >= n) return;
  float a0 = a[i], a1 = a[(size_t)n + i], a2 = a[2 * (size_t)n + i];
  float b0 = b[i], b1 = b[(size_t)n + i], b2 = b[2 * (size_t)n + i];
  x[i]                 = a1 * b2 - a2 * b1;
  x[(size_t)n + i]     = a2 * b0 - a0 * b2;
  x[2 * (size_t)n + i] = a0 * b1 - a1 * b0;
}
// X_{ij} = sum_k L_{ik} R_{kj}
__global__ void k_mat3(float* __restrict__ x, const float* __restrict__ L,
                       const float* __restrict__ R, int n) {
  int i = blockIdx.x * 256 + threadIdx.x;
  if (i >= n) return;
  float l[9], r[9];
#pragma unroll
  for (int q = 0; q < 9; ++q) { l[q] = L[(size_t)q * n + i]; r[q] = R[(size_t)q * n + i]; }
#pragma unroll
  for (int a = 0; a < 3; ++a)
#pragma unroll
    for (int bb = 0; bb < 3; ++bb)
      x[(size_t)(a * 3 + bb) * n + i] =
          l[a * 3 + 0] * r[0 * 3 + bb] + l[a * 3 + 1] * r[1 * 3 + bb] + l[a * 3 + 2] * r[2 * 3 + bb];
}
// group norms (residual fused: base += GN(dx))
__global__ void k_gn_a(float* __restrict__ base, const float* __restrict__ dx,
                       const float* __restrict__ gamma, const float* __restrict__ beta) {
  int idx = blockIdx.x * 256 + threadIdx.x;
  if (idx >= P_TOT * 8) return;
  int p = idx >> 3, g = idx & 7;
  const float* x = dx + (size_t)p * 256 + g * 32;
  float mean = 0.f;
  for (int c = 0; c < 32; ++c) mean += x[c];
  mean *= (1.f / 32.f);
  float var = 0.f;
  for (int c = 0; c < 32; ++c) { float d = x[c] - mean; var += d * d; }
  var *= (1.f / 32.f);
  float inv = rsqrtf(var + 1e-5f);
  float* o = base + (size_t)p * 256 + g * 32;
  for (int c = 0; c < 32; ++c) {
    int ch = g * 32 + c;
    o[c] += (x[c] - mean) * inv * gamma[ch] + beta[ch];
  }
}
__global__ void k_gn_v(float* __restrict__ base, const float* __restrict__ dx,
                       const float* __restrict__ gamma) {
  int idx = blockIdx.x * 256 + threadIdx.x;
  if (idx >= P_TOT * 8) return;
  int p = idx >> 3, g = idx & 7;
  float ss = 0.f;
  for (int j = 0; j < 3; ++j)
    for (int c = 0; c < 16; ++c) {
      float v = dx[(size_t)j * PV_ + (size_t)p * 128 + g * 16 + c];
      ss += v * v;
    }
  float inv = rsqrtf(ss * (1.f / 48.f) + 1e-5f);
  for (int j = 0; j < 3; ++j)
    for (int c = 0; c < 16; ++c) {
      size_t ix = (size_t)j * PV_ + (size_t)p * 128 + g * 16 + c;
      base[ix] += dx[ix] * inv * gamma[g * 16 + c];
    }
}
__global__ void k_gn_d(float* __restrict__ base, const float* __restrict__ dx,
                       const float* __restrict__ gamma) {
  int idx = blockIdx.x * 256 + threadIdx.x;
  if (idx >= P_TOT * 8) return;
  int p = idx >> 3, g = idx & 7;
  float ss = 0.f;
  for (int j = 0; j < 9; ++j)
    for (int c = 0; c < 8; ++c) {
      float v = dx[(size_t)j * PD_ + (size_t)p * 64 + g * 8 + c];
      ss += v * v;
    }
  float inv = rsqrtf(ss * (1.f / 72.f) + 1e-5f);
  for (int j = 0; j < 9; ++j)
    for (int c = 0; c < 8; ++c) {
      size_t ix = (size_t)j * PD_ + (size_t)p * 64 + g * 8 + c;
      base[ix] += dx[ix] * inv * gamma[g * 8 + c];
    }
}
// equivariant gate: x *= sigmoid(||x||_F over comps planes)
__global__ void k_gate(float* __restrict__ x, int n, int comps) {
  int i = blockIdx.x * 256 + threadIdx.x;
  if (i >= n) return;
  float ss = 0.f;
  for (int t = 0; t < comps; ++t) { float v = x[(size_t)t * n + i]; ss += v * v; }
  float nrm = sqrtf(ss + 1e-12f);
  float s = 1.f / (1.f + expf(-nrm));
  for (int t = 0; t < comps; ++t) x[(size_t)t * n + i] *= s;
}
__global__ void k_leaky(float* __restrict__ x, int n) {
  int i = blockIdx.x * 256 + threadIdx.x;
  if (i < n) { float v = x[i]; x[i] = v > 0.f ? v : 0.1f * v; }
}
// final residual combines (+ re-interleave for v/d)
__global__ void k_out_a(float* __restrict__ o, const float* __restrict__ xa,
                        const float* __restrict__ mo, const float* __restrict__ co, int n) {
  int i = blockIdx.x * 256 + threadIdx.x;
  if (i < n) o[i] = xa[i] + mo[i] + co[i];
}
__global__ void k_out_v(float* __restrict__ o, const float* __restrict__ xv,
                        const float* __restrict__ mo, const float* __restrict__ co) {
  int i = blockIdx.x * 256 + threadIdx.x;
  if (i >= PV_) return;
#pragma unroll
  for (int j = 0; j < 3; ++j)
    o[3 * (size_t)i + j] = xv[3 * (size_t)i + j] + mo[(size_t)j * PV_ + i] + co[(size_t)j * PV_ + i];
}
__global__ void k_out_d(float* __restrict__ o, const float* __restrict__ xd,
                        const float* __restrict__ mo, const float* __restrict__ co) {
  int i = blockIdx.x * 256 + threadIdx.x;
  if (i >= PD_) return;
#pragma unroll
  for (int q = 0; q < 9; ++q)
    o[9 * (size_t)i + q] = xd[9 * (size_t)i + q] + mo[(size_t)q * PD_ + i] + co[(size_t)q * PD_ + i];
}

// ------------------------------- host side ---------------------------------
static inline int g1(int n) { return (n + 255) / 256; }

static inline void gemm(hipStream_t s, const float* A, int lda,
                        const float* W, int ldw,
                        float* C, int ldc, int N, int K,
                        float alpha, int accum, int shift, int act) {
  const int tiles = (P_TOT / 32) * (N / 32);
  const int blocks = (tiles + 7) / 8;
  switch (K) {
    case 64:
      k_wmma_gemm<64><<<blocks, 256, 0, s>>>(A, lda, W, ldw, C, ldc, N, alpha, accum, shift, act);
      break;
    case 128:
      k_wmma_gemm<128><<<blocks, 256, 0, s>>>(A, lda, W, ldw, C, ldc, N, alpha, accum, shift, act);
      break;
    default:
      k_wmma_gemm<256><<<blocks, 256, 0, s>>>(A, lda, W, ldw, C, ldc, N, alpha, accum, shift, act);
      break;
  }
}

extern "C" void kernel_launch(void* const* d_in, const int* in_sizes, int n_in,
                              void* d_out, int out_size, void* d_ws, size_t ws_size,
                              hipStream_t stream) {
  (void)in_sizes; (void)n_in; (void)out_size; (void)ws_size;
  auto F = [&](int i) { return (const float*)d_in[i]; };
  const float* x_a = F(0); const float* x_v = F(1); const float* x_d = F(2);
  const float* Waa = F(3), *Wad = F(4), *Wvv = F(5), *Wvd = F(6);
  const float* Wdd = F(7), *Wdt = F(8), *Wda = F(9), *Wdv = F(10);
  const float* TPL[9], *TPR[9], *TPO[9];
  for (int q = 0; q < 9; ++q) { TPL[q] = F(11 + 3 * q); TPR[q] = F(12 + 3 * q); TPO[q] = F(13 + 3 * q); }
  enum { AA = 0, VV, DDp, AV, DVp, VXV, AD, VOV, DMD };
  const float* g_ag = F(38), *g_ab = F(39), *g_vg = F(40), *g_dg = F(41);
  const float* conv_a[3] = { F(42), F(43), F(44) };
  const float* conv_v[3] = { F(45), F(46), F(47) };
  const float* conv_d[3] = { F(48), F(49), F(50) };
  const float* mlp_a[3]  = { F(51), F(52), F(53) };
  const float* mlp_v[3]  = { F(54), F(55), F(56) };
  const float* mlp_d[3]  = { F(57), F(58), F(59) };

  // ---- workspace arena (fp32; rank temps reused as chain temps)
  float* w = (float*)d_ws; size_t off = 0;
  auto alloc = [&](size_t n) { float* p = w + off; off += n; return p; };
  float* vpl = alloc(3 * (size_t)PV_);   // input v planes
  float* dpl = alloc(9 * (size_t)PD_);   // input d planes
  float* trb = alloc((size_t)PD_);       // trace(d)
  float* vdp = alloc(3 * (size_t)PD_);   // eps.d planes
  float* a2  = alloc((size_t)PA_);       // mixed a  (becomes post-GN stream)
  float* v2  = alloc(3 * (size_t)PV_);   // mixed v planes
  float* d2  = alloc(9 * (size_t)PD_);   // mixed d planes
  float* tpl = alloc(3 * (size_t)PD_);   // Wdv.v_k
  float* da  = alloc((size_t)PA_);       // tp out (a)
  float* dvb = alloc(3 * (size_t)PV_);   // tp out (v planes)
  float* ddb = alloc(9 * (size_t)PD_);   // tp out (d planes)
  float* Lb  = alloc(9 * (size_t)PD_);   // rank temps (L)
  float* Rb  = alloc(9 * (size_t)PD_);   // rank temps (R)
  float* Xb  = alloc(9 * (size_t)PD_);   // rank temps (combined)
  float* co  = alloc(9 * (size_t)PD_);   // conv chain output
  float* cwa = alloc(3 * 7 * (size_t)DA_ * DA_);   // split conv weights (a)
  float* cwv = alloc(3 * 7 * (size_t)DV_ * DV_);   // split conv weights (v)
  float* cwd = alloc(3 * 7 * (size_t)DD_ * DD_);   // split conv weights (d)
  float *t0 = Lb, *t1 = Rb, *mo = Xb;    // chain temps alias rank temps

  float* out_a = (float*)d_out;
  float* out_v = out_a + (size_t)PA_;
  float* out_d = out_v + 3 * (size_t)PV_;

  // ---- phase 0: de-interleave inputs + split conv kernels into per-tap mats
  k_split_v<<<g1(PV_), 256, 0, stream>>>(x_v, vpl);
  k_split_d<<<g1(PD_), 256, 0, stream>>>(x_d, dpl, trb, vdp);
  for (int l = 0; l < 3; ++l) {
    k_split_conv<<<g1(DA_ * DA_ * 7), 256, 0, stream>>>(conv_a[l], cwa + (size_t)l * 7 * DA_ * DA_, DA_ * DA_);
    k_split_conv<<<g1(DV_ * DV_ * 7), 256, 0, stream>>>(conv_v[l], cwv + (size_t)l * 7 * DV_ * DV_, DV_ * DV_);
    k_split_conv<<<g1(DD_ * DD_ * 7), 256, 0, stream>>>(conv_d[l], cwd + (size_t)l * 7 * DD_ * DD_, DD_ * DD_);
  }

  // ---- phase 1: linear_mix  (all WMMA GEMMs)
  gemm(stream, x_a, DA_, Waa, DA_, a2, DA_, DA_, DA_, 1.f, 0, 0, 0);
  gemm(stream, trb, DD_, Wad, DD_, a2, DA_, DA_, DD_, 1.f, 1, 0, 0);
  for (int j = 0; j < 3; ++j) {
    gemm(stream, vpl + (size_t)j * PV_, DV_, Wvv, DV_, v2 + (size_t)j * PV_, DV_, DV_, DV_, 1.f, 0, 0, 0);
    gemm(stream, vdp + (size_t)j * PD_, DD_, Wvd, DD_, v2 + (size_t)j * PV_, DV_, DV_, DD_, 1.f, 1, 0, 0);
  }
  for (int ij = 0; ij < 9; ++ij) {
    int i = ij / 3, j = ij % 3, ji = j * 3 + i;
    gemm(stream, dpl + (size_t)ij * PD_, DD_, Wdd, DD_, d2 + (size_t)ij * PD_, DD_, DD_, DD_, 1.f, 0, 0, 0);
    gemm(stream, dpl + (size_t)ji * PD_, DD_, Wdt, DD_, d2 + (size_t)ij * PD_, DD_, DD_, DD_, 1.f, 1, 0, 0);
    if (i == j)
      gemm(stream, x_a, DA_, Wda, DA_, d2 + (size_t)ij * PD_, DD_, DD_, DA_, 1.f, 1, 0, 0);
  }
  for (int k = 0; k < 3; ++k)
    gemm(stream, vpl + (size_t)k * PV_, DV_, Wdv, DV_, tpl + (size_t)k * PD_, DD_, DD_, DV_, 1.f, 0, 0, 0);
  k_eps_add<<<g1(PD_), 256, 0, stream>>>(d2, tpl);

  // ---- phase 2: rank-64 tensor products (0.3 scaling folded into l/r GEMMs)
  const float S = 0.3f;
  // aa / vv / dd -> da
  gemm(stream, a2, DA_, TPL[AA], DA_, Lb, DD_, DD_, DA_, S, 0, 0, 0);
  gemm(stream, a2, DA_, TPR[AA], DA_, Rb, DD_, DD_, DA_, S, 0, 0, 0);
  k_had<<<g1(PD_), 256, 0, stream>>>(Xb, Lb, Rb, PD_);
  gemm(stream, Xb, DD_, TPO[AA], DD_, da, DA_, DA_, DD_, 1.f, 0, 0, 0);
  for (int j = 0; j < 3; ++j) {
    gemm(stream, v2 + (size_t)j * PV_, DV_, TPL[VV], DV_, Lb + (size_t)j * PD_, DD_, DD_, DV_, S, 0, 0, 0);
    gemm(stream, v2 + (size_t)j * PV_, DV_, TPR[VV], DV_, Rb + (size_t)j * PD_, DD_, DD_, DV_, S, 0, 0, 0);
  }
  k_dotn<<<g1(PD_), 256, 0, stream>>>(Xb, Lb, Rb, PD_, 3);
  gemm(stream, Xb, DD_, TPO[VV], DD_, da, DA_, DA_, DD_, 1.f, 1, 0, 0);
  for (int q = 0; q < 9; ++q) {
    gemm(stream, d2 + (size_t)q * PD_, DD_, TPL[DDp], DD_, Lb + (size_t)q * PD_, DD_, DD_, DD_, S, 0, 0, 0);
    gemm(stream, d2 + (size_t)q * PD_, DD_, TPR[DDp], DD_, Rb + (size_t)q * PD_, DD_, DD_, DD_, S, 0, 0, 0);
  }
  k_dotn<<<g1(PD_), 256, 0, stream>>>(Xb, Lb, Rb, PD_, 9);
  gemm(stream, Xb, DD_, TPO[DDp], DD_, da, DA_, DA_, DD_, 1.f, 1, 0, 0);
  // av / dv / vxv -> dvb planes
  gemm(stream, a2, DA_, TPL[AV], DA_, Lb, DD_, DD_, DA_, S, 0, 0, 0);
  for (int j = 0; j < 3; ++j)
    gemm(stream, v2 + (size_t)j * PV_, DV_, TPR[AV], DV_, Rb + (size_t)j * PD_, DD_, DD_, DV_, S, 0, 0, 0);
  for (int j = 0; j < 3; ++j) {
    k_had<<<g1(PD_), 256, 0, stream>>>(Xb + (size_t)j * PD_, Lb, Rb + (size_t)j * PD_, PD_);
    gemm(stream, Xb + (size_t)j * PD_, DD_, TPO[AV], DD_, dvb + (size_t)j * PV_, DV_, DV_, DD_, 1.f, 0, 0, 0);
  }
  for (int q = 0; q < 9; ++q)
    gemm(stream, d2 + (size_t)q * PD_, DD_, TPL[DVp], DD_, Lb + (size_t)q * PD_, DD_, DD_, DD_, S, 0, 0, 0);
  for (int j = 0; j < 3; ++j)
    gemm(stream, v2 + (size_t)j * PV_, DV_, TPR[DVp], DV_, Rb + (size_t)j * PD_, DD_, DD_, DV_, S, 0, 0, 0);
  k_matvec3<<<g1(PD_), 256, 0, stream>>>(Xb, Lb, Rb, PD_);
  for (int j = 0; j < 3; ++j)
    gemm(stream, Xb + (size_t)j * PD_, DD_, TPO[DVp], DD_, dvb + (size_t)j * PV_, DV_, DV_, DD_, 1.f, 1, 0, 0);
  for (int j = 0; j < 3; ++j) {
    gemm(stream, v2 + (size_t)j * PV_, DV_, TPL[VXV], DV_, Lb + (size_t)j * PD_, DD_, DD_, DV_, S, 0, 0, 0);
    gemm(stream, v2 + (size_t)j * PV_, DV_, TPR[VXV], DV_, Rb + (size_t)j * PD_, DD_, DD_, DV_, S, 0, 0, 0);
  }
  k_cross3<<<g1(PD_), 256, 0, stream>>>(Xb, Lb, Rb, PD_);
  for (int j = 0; j < 3; ++j)
    gemm(stream, Xb + (size_t)j * PD_, DD_, TPO[VXV], DD_, dvb + (size_t)j * PV_, DV_, DV_, DD_, 1.f, 1, 0, 0);
  // ad / vov / dmd -> ddb planes
  gemm(stream, a2, DA_, TPL[AD], DA_, Lb, DD_, DD_, DA_, S, 0, 0, 0);
  for (int q = 0; q < 9; ++q) {
    gemm(stream, d2 + (size_t)q * PD_, DD_, TPR[AD], DD_, Rb + (size_t)q * PD_, DD_, DD_, DD_, S, 0, 0, 0);
    k_had<<<g1(PD_), 256, 0, stream>>>(Xb + (size_t)q * PD_, Lb, Rb + (size_t)q * PD_, PD_);
    gemm(stream, Xb + (size_t)q * PD_, DD_, TPO[AD], DD_, ddb + (size_t)q * PD_, DD_, DD_, DD_, 1.f, 0, 0, 0);
  }
  for (int j = 0; j < 3; ++j) {
    gemm(stream, v2 + (size_t)j * PV_, DV_, TPL[VOV], DV_, Lb + (size_t)j * PD_, DD_, DD_, DV_, S, 0, 0, 0);
    gemm(stream, v2 + (size_t)j * PV_, DV_, TPR[VOV], DV_, Rb + (size_t)j * PD_, DD_, DD_, DV_, S, 0, 0, 0);
  }
  for (int q = 0; q < 9; ++q) {
    k_had<<<g1(PD_), 256, 0, stream>>>(Xb + (size_t)q * PD_, Lb + (size_t)(q / 3) * PD_, Rb + (size_t)(q % 3) * PD_, PD_);
    gemm(stream, Xb + (size_t)q * PD_, DD_, TPO[VOV], DD_, ddb + (size_t)q * PD_, DD_, DD_, DD_, 1.f, 1, 0, 0);
  }
  for (int q = 0; q < 9; ++q) {
    gemm(stream, d2 + (size_t)q * PD_, DD_, TPL[DMD], DD_, Lb + (size_t)q * PD_, DD_, DD_, DD_, S, 0, 0, 0);
    gemm(stream, d2 + (size_t)q * PD_, DD_, TPR[DMD], DD_, Rb + (size_t)q * PD_, DD_, DD_, DD_, S, 0, 0, 0);
  }
  k_mat3<<<g1(PD_), 256, 0, stream>>>(Xb, Lb, Rb, PD_);
  for (int q = 0; q < 9; ++q)
    gemm(stream, Xb + (size_t)q * PD_, DD_, TPO[DMD], DD_, ddb + (size_t)q * PD_, DD_, DD_, DD_, 1.f, 1, 0, 0);

  // ---- phase 3: group norm + residual (in place into a2/v2/d2 streams)
  k_gn_a<<<g1(P_TOT * 8), 256, 0, stream>>>(a2, da, g_ag, g_ab);
  k_gn_v<<<g1(P_TOT * 8), 256, 0, stream>>>(v2, dvb, g_vg);
  k_gn_d<<<g1(P_TOT * 8), 256, 0, stream>>>(d2, ddb, g_dg);

  // ---- phase 4: MLP + chain-conv per stream, then combine
  auto conv_layer = [&](const float* src, float* dst, const float* Wt /*7 mats C*C*/, int C) {
    for (int t = 0; t < 7; ++t)
      gemm(stream, src, C, Wt + (size_t)t * C * C, C, dst, C, C, C, 1.f, t > 0, t - 3, 0);
  };
  // a-stream (leaky fused in MLP; separate for conv post-sum)
  gemm(stream, a2, DA_, mlp_a[0], DA_, t0, DA_, DA_, DA_, 1.f, 0, 0, 1);
  gemm(stream, t0, DA_, mlp_a[1], DA_, t1, DA_, DA_, DA_, 1.f, 0, 0, 1);
  gemm(stream, t1, DA_, mlp_a[2], DA_, mo, DA_, DA_, DA_, 1.f, 0, 0, 0);
  conv_layer(a2, t0, cwa + (size_t)0 * 7 * DA_ * DA_, DA_); k_leaky<<<g1(PA_), 256, 0, stream>>>(t0, PA_);
  conv_layer(t0, t1, cwa + (size_t)1 * 7 * DA_ * DA_, DA_); k_leaky<<<g1(PA_), 256, 0, stream>>>(t1, PA_);
  conv_layer(t1, co, cwa + (size_t)2 * 7 * DA_ * DA_, DA_);
  k_out_a<<<g1(PA_), 256, 0, stream>>>(out_a, x_a, mo, co, PA_);
  // v-stream
  for (int j = 0; j < 3; ++j)
    gemm(stream, v2 + (size_t)j * PV_, DV_, mlp_v[0], DV_, t0 + (size_t)j * PV_, DV_, DV_, DV_, 1.f, 0, 0, 0);
  k_gate<<<g1(PV_), 256, 0, stream>>>(t0, PV_, 3);
  for (int j = 0; j < 3; ++j)
    gemm(stream, t0 + (size_t)j * PV_, DV_, mlp_v[1], DV_, t1 + (size_t)j * PV_, DV_, DV_, DV_, 1.f, 0, 0, 0);
  k_gate<<<g1(PV_), 256, 0, stream>>>(t1, PV_, 3);
  for (int j = 0; j < 3; ++j)
    gemm(stream, t1 + (size_t)j * PV_, DV_, mlp_v[2], DV_, mo + (size_t)j * PV_, DV_, DV_, DV_, 1.f, 0, 0, 0);
  for (int j = 0; j < 3; ++j) conv_layer(v2 + (size_t)j * PV_, t0 + (size_t)j * PV_, cwv + (size_t)0 * 7 * DV_ * DV_, DV_);
  k_gate<<<g1(PV_), 256, 0, stream>>>(t0, PV_, 3);
  for (int j = 0; j < 3; ++j) conv_layer(t0 + (size_t)j * PV_, t1 + (size_t)j * PV_, cwv + (size_t)1 * 7 * DV_ * DV_, DV_);
  k_gate<<<g1(PV_), 256, 0, stream>>>(t1, PV_, 3);
  for (int j = 0; j < 3; ++j) conv_layer(t1 + (size_t)j * PV_, co + (size_t)j * PV_, cwv + (size_t)2 * 7 * DV_ * DV_, DV_);
  k_out_v<<<g1(PV_), 256, 0, stream>>>(out_v, x_v, mo, co);
  // d-stream
  for (int q = 0; q < 9; ++q)
    gemm(stream, d2 + (size_t)q * PD_, DD_, mlp_d[0], DD_, t0 + (size_t)q * PD_, DD_, DD_, DD_, 1.f, 0, 0, 0);
  k_gate<<<g1(PD_), 256, 0, stream>>>(t0, PD_, 9);
  for (int q = 0; q < 9; ++q)
    gemm(stream, t0 + (size_t)q * PD_, DD_, mlp_d[1], DD_, t1 + (size_t)q * PD_, DD_, DD_, DD_, 1.f, 0, 0, 0);
  k_gate<<<g1(PD_), 256, 0, stream>>>(t1, PD_, 9);
  for (int q = 0; q < 9; ++q)
    gemm(stream, t1 + (size_t)q * PD_, DD_, mlp_d[2], DD_, mo + (size_t)q * PD_, DD_, DD_, DD_, 1.f, 0, 0, 0);
  for (int q = 0; q < 9; ++q) conv_layer(d2 + (size_t)q * PD_, t0 + (size_t)q * PD_, cwd + (size_t)0 * 7 * DD_ * DD_, DD_);
  k_gate<<<g1(PD_), 256, 0, stream>>>(t0, PD_, 9);
  for (int q = 0; q < 9; ++q) conv_layer(t0 + (size_t)q * PD_, t1 + (size_t)q * PD_, cwd + (size_t)1 * 7 * DD_ * DD_, DD_);
  k_gate<<<g1(PD_), 256, 0, stream>>>(t1, PD_, 9);
  for (int q = 0; q < 9; ++q) conv_layer(t1 + (size_t)q * PD_, co + (size_t)q * PD_, cwd + (size_t)2 * 7 * DD_ * DD_, DD_);
  k_out_d<<<g1(PD_), 256, 0, stream>>>(out_d, x_d, mo, co);
}
